// Block_45397804319450
// MI455X (gfx1250) — matrix operand order repaired and checked by reading, not compiled, not run on hardware
//
#include <hip/hip_runtime.h>
#include <hip/hip_bf16.h>

// ---------------------------------------------------------------------------
// Types
// ---------------------------------------------------------------------------
typedef __bf16 bf16;
typedef __bf16    v16bf __attribute__((ext_vector_type(16)));
typedef __bf16    v8bf  __attribute__((ext_vector_type(8)));
typedef float     v8f   __attribute__((ext_vector_type(8)));
typedef unsigned  v4u   __attribute__((ext_vector_type(4)));
typedef int       v8i   __attribute__((ext_vector_type(8)));
typedef int       v4i   __attribute__((ext_vector_type(4)));

#define HAVE_TDM (__has_builtin(__builtin_amdgcn_tensor_load_to_lds) && \
                  __has_builtin(__builtin_amdgcn_s_wait_tensorcnt))

__device__ __forceinline__ bf16 f2bf(float f) {
    unsigned u = __float_as_uint(f);
    unsigned short s = (unsigned short)((u + 0x7FFFu + ((u >> 16) & 1u)) >> 16);
    return __builtin_bit_cast(bf16, s);
}

#define GMODE_QKV  0
#define GMODE_RES  1
#define GMODE_GELU 2

// ---------------------------------------------------------------------------
// TDM: 2D tile load, 128 rows x 32 bf16, row stride = strideElems,
// LDS pitch padded to 40 elems (pad 4 DWORDs after every 16 DWORDs).
// D# packing per CDNA5 ISA.
// 6-arg builtin flavor: (u32x4 g0, i32x8 g1, i32x4, i32x4, i32x8, i32 cpol)
// ---------------------------------------------------------------------------
__device__ __forceinline__ void tdm_load_tile_128x32(unsigned lds_off,
                                                     const bf16* g,
                                                     unsigned strideElems) {
#if HAVE_TDM
    unsigned long long ga = (unsigned long long)(const void*)g;
    v4u g0;
    g0.x = 1u;                                       // count=1, user mode
    g0.y = lds_off;                                  // lds_addr (bytes)
    g0.z = (unsigned)ga;                             // global_addr[31:0]
    g0.w = (unsigned)((ga >> 32) & 0x01FFFFFFu) | 0x80000000u;  // addr[56:32] | type=2
    v8i g1;
    g1[0] = (int)((1u << 16)      // data_size = 2 bytes
                | (1u << 20)      // pad_enable
                | (3u << 22)      // pad_interval: 16 DWORDs (64B rows)
                | (3u << 25));    // pad_amount:   4 DWORDs (16B -> pitch 80B)
    unsigned td0 = strideElems;                      // tensor_dim0
    g1[1] = (int)((td0 & 0xFFFFu) << 16);            // dim0[15:0]
    g1[2] = (int)(((td0 >> 16) & 0xFFFFu) | (128u << 16)); // dim0[31:16] | dim1=128
    g1[3] = (int)(32u << 16);                        // dim1 hi=0 | tile_dim0=32
    g1[4] = (int)128u;                               // tile_dim1=128, tile_dim2=0
    g1[5] = (int)strideElems;                        // dim0_stride[31:0]
    g1[6] = 0;                                       // dim0_stride hi, dim1_stride lo
    g1[7] = 0;
    v4i z4 = {0, 0, 0, 0};
    v8i z8 = {0, 0, 0, 0, 0, 0, 0, 0};
    __builtin_amdgcn_tensor_load_to_lds(g0, g1, z4, z4, z8, 0);
#else
    (void)lds_off; (void)g; (void)strideElems;
#endif
}

// ---------------------------------------------------------------------------
// Weight prep: fp32 [K,N] -> bf16 transposed [N,K]
// ---------------------------------------------------------------------------
__global__ __launch_bounds__(256)
void transpose_bf16_kernel(const float* __restrict__ in, bf16* __restrict__ out,
                           int K, int N) {
    __shared__ float tile[32][33];
    int kb = blockIdx.y * 32, nb = blockIdx.x * 32;
    int tx = threadIdx.x & 31, ty = threadIdx.x >> 5;   // 32 x 8
    #pragma unroll
    for (int i = ty; i < 32; i += 8)
        tile[i][tx] = in[(size_t)(kb + i) * N + nb + tx];
    __syncthreads();
    #pragma unroll
    for (int i = ty; i < 32; i += 8)
        out[(size_t)(nb + i) * K + kb + tx] = f2bf(tile[tx][i]);
}

// ---------------------------------------------------------------------------
// LayerNorm: fp32 [rows, d] -> bf16 [rows, d]
// ---------------------------------------------------------------------------
__global__ __launch_bounds__(256)
void ln_bf16_kernel(const float* __restrict__ x, const float* __restrict__ g,
                    const float* __restrict__ bta, bf16* __restrict__ out, int d) {
    int row = blockIdx.x;
    const float* xr = x + (size_t)row * d;
    __shared__ float red[2][8];
    float s = 0.f, s2 = 0.f;
    for (int i = threadIdx.x; i < d; i += 256) { float v = xr[i]; s += v; s2 += v * v; }
    for (int off = 16; off > 0; off >>= 1) {
        s  += __shfl_xor(s,  off, 32);
        s2 += __shfl_xor(s2, off, 32);
    }
    int wv = threadIdx.x >> 5, lane = threadIdx.x & 31;
    if (lane == 0) { red[0][wv] = s; red[1][wv] = s2; }
    __syncthreads();
    s = 0.f; s2 = 0.f;
    #pragma unroll
    for (int i = 0; i < 8; ++i) { s += red[0][i]; s2 += red[1][i]; }
    float mu  = s / (float)d;
    float var = s2 / (float)d - mu * mu;
    float rs  = rsqrtf(var + 1e-5f);
    for (int i = threadIdx.x; i < d; i += 256) {
        float v = (xr[i] - mu) * rs * g[i] + bta[i];
        out[(size_t)row * d + i] = f2bf(v);
    }
}

// ---------------------------------------------------------------------------
// Tiled bf16 WMMA GEMM: C[M,N] = A[M,K] * Bt[N,K]^T + bias, fused epilogues.
// Block tile 128x128, BK=32, 128 threads = 4 waves, wave tile 64x64.
// Tiles staged to (dynamic) LDS by the Tensor Data Mover, double buffered.
// Dynamic LDS + pointer-escape asm keep the compiler from proving the tile
// buffer "never written" (the TDM write is invisible to alias analysis).
// ---------------------------------------------------------------------------
__global__ __launch_bounds__(128)
void gemm_bf16_kernel(const bf16* __restrict__ A, const bf16* __restrict__ Bt,
                      const float* __restrict__ bias,
                      int M, int N, int K, int mode,
                      const float* __restrict__ res, float* __restrict__ outf,
                      bf16* __restrict__ outb,
                      bf16* __restrict__ qo, bf16* __restrict__ ko, bf16* __restrict__ vto,
                      int Tdim, int Hheads) {
    // [A0 | A1 | B0 | B1], each 128 rows * 40 elems (80B pitch, 16B aligned runs)
    extern __shared__ __align__(16) bf16 smem[];
    const int TILE_E = 128 * 40;           // 5120 elems, 10240 bytes

    int tid  = threadIdx.x;
    int lane = tid & 31;
    int wv   = tid >> 5;
    int half = lane >> 4;
    int l16  = lane & 15;
    int wrow = (wv & 1) * 64;              // wave M offset in block tile
    int wcol = (wv >> 1) * 64;             // wave N offset
    int mtile = blockIdx.y * 128;
    int ntile = blockIdx.x * 128;

    v8f zero8 = {0.f, 0.f, 0.f, 0.f, 0.f, 0.f, 0.f, 0.f};
    v8f acc[4][4];
    #pragma unroll
    for (int mi = 0; mi < 4; ++mi)
        #pragma unroll
        for (int ni = 0; ni < 4; ++ni) acc[mi][ni] = zero8;

    int nk = K >> 5;

#if HAVE_TDM
    if (wv == 0) {
        tdm_load_tile_128x32(0u,      A  + (size_t)mtile * K, (unsigned)K);
        tdm_load_tile_128x32(20480u,  Bt + (size_t)ntile * K, (unsigned)K);
    }
#endif

    for (int ki = 0; ki < nk; ++ki) {
        int buf = ki & 1;
        const bf16* Asm;
        const bf16* Bsm;
#if HAVE_TDM
        if (wv == 0) __builtin_amdgcn_s_wait_tensorcnt(0);
        __syncthreads();                   // tiles(ki) visible; buf^1 free
        // The TDM wrote smem behind the compiler's back: escape the pointer so
        // the ds_load fragment reads below cannot be folded away.
        {
            bf16* esc = smem;
            asm volatile("" : "+v"(esc) :: "memory");
        }
        if (wv == 0 && ki + 1 < nk) {
            tdm_load_tile_128x32((unsigned)((ki + 1) & 1) * 10240u,
                                 A + (size_t)mtile * K + (size_t)(ki + 1) * 32,
                                 (unsigned)K);
            tdm_load_tile_128x32(20480u + (unsigned)((ki + 1) & 1) * 10240u,
                                 Bt + (size_t)ntile * K + (size_t)(ki + 1) * 32,
                                 (unsigned)K);
        }
        Asm = smem + buf * TILE_E;
        Bsm = smem + (2 + buf) * TILE_E;
#else
        // ---- manual staging: both tiles are straight [row][32] copies ----
        int k0 = ki * 32;
        bf16* Aw = smem;                   // single buffer
        bf16* Bw = smem + 2 * TILE_E;
        #pragma unroll
        for (int i = 0; i < 4; ++i) {
            int c   = i * 128 + tid;       // 0..511 chunks of 8 bf16
            int row = c >> 2;
            int col = (c & 3) * 8;
            *(v8bf*)&Aw[row * 40 + col] =
                *(const v8bf*)&A[(size_t)(mtile + row) * K + k0 + col];
            *(v8bf*)&Bw[row * 40 + col] =
                *(const v8bf*)&Bt[(size_t)(ntile + row) * K + k0 + col];
        }
        if (ki + 1 < nk) {
            __builtin_prefetch(&A[(size_t)(mtile + tid) * K + k0 + 32], 0, 0);
            __builtin_prefetch(&Bt[(size_t)(ntile + tid) * K + k0 + 32], 0, 0);
        }
        __syncthreads();
        Asm = Aw;
        Bsm = Bw;
#endif

        // ---- A fragments (lane=M; half h holds contiguous K run [h*16,+16)) ----
        v16bf af[4];
        #pragma unroll
        for (int mi = 0; mi < 4; ++mi) {
            int row = wrow + mi * 16 + l16;
            int kb  = half * 16;
            union { v16bf v; v8bf h[2]; } u;
            u.h[0] = *(const v8bf*)&Asm[row * 40 + kb];
            u.h[1] = *(const v8bf*)&Asm[row * 40 + kb + 8];
            af[mi] = u.v;
        }
        // ---- B fragments (lane=N; half h holds contiguous K run [h*16,+16)) ----
        v16bf bfm[4];
        #pragma unroll
        for (int ni = 0; ni < 4; ++ni) {
            int nn = wcol + ni * 16 + l16;
            int kb = half * 16;
            union { v16bf v; v8bf h[2]; } u;
            u.h[0] = *(const v8bf*)&Bsm[nn * 40 + kb];
            u.h[1] = *(const v8bf*)&Bsm[nn * 40 + kb + 8];
            bfm[ni] = u.v;
        }
        #pragma unroll
        for (int mi = 0; mi < 4; ++mi)
            #pragma unroll
            for (int ni = 0; ni < 4; ++ni)
                acc[mi][ni] = __builtin_amdgcn_wmma_f32_16x16x32_bf16(
                    false, af[mi], false, bfm[ni], (short)0, acc[mi][ni], false, false);
#if !HAVE_TDM
        __syncthreads();
#endif
    }

    // ---- epilogue (C layout: lane=N%16, VGPR r => M = r + 8*half) ----
    #pragma unroll
    for (int mi = 0; mi < 4; ++mi) {
        #pragma unroll
        for (int ni = 0; ni < 4; ++ni) {
            int n = ntile + wcol + ni * 16 + l16;
            float bv = bias[n];
            #pragma unroll
            for (int r = 0; r < 8; ++r) {
                int m = mtile + wrow + mi * 16 + r + half * 8;
                float v = acc[mi][ni][r] + bv;
                if (mode == GMODE_RES) {
                    outf[(size_t)m * N + n] = v + res[(size_t)m * N + n];
                } else if (mode == GMODE_GELU) {
                    float gl = 0.5f * v * (1.f + erff(v * 0.70710678f));
                    outb[(size_t)m * N + n] = f2bf(gl);
                } else { // GMODE_QKV: scatter into Q/K [B,H,T,64], V transposed [B,H,64,T]
                    int comp = n >> 10;          // 0=q 1=k 2=v
                    int cd   = n & 1023;
                    int hh   = cd >> 6;
                    int dk   = cd & 63;
                    int b    = m / Tdim;
                    int t    = m - b * Tdim;
                    size_t bh = (size_t)(b * Hheads + hh);
                    bf16 bvv = f2bf(v);
                    if (comp == 0)      qo[(bh * Tdim + t) * 64 + dk] = bvv;
                    else if (comp == 1) ko[(bh * Tdim + t) * 64 + dk] = bvv;
                    else                vto[(bh * 64 + dk) * Tdim + t] = bvv;
                }
            }
        }
    }
}

// ---------------------------------------------------------------------------
// Flash attention: one wave per 16-query tile, key tiles of 32, online softmax.
// Q,K: bf16 [BH, T, 64]; Vt: bf16 [BH, 64, T]; out: bf16 [B*T, H*64].
// ---------------------------------------------------------------------------
__global__ __launch_bounds__(128)
void attn_kernel(const bf16* __restrict__ Qg, const bf16* __restrict__ Kg,
                 const bf16* __restrict__ Vt, bf16* __restrict__ out,
                 int H, int T, int BH) {
    __shared__ __align__(16) bf16 Pst[4][16][32];  // per-wave P staging [M][K]

    int tid  = threadIdx.x;
    int lane = tid & 31;
    int wv   = tid >> 5;
    int half = lane >> 4;
    int l16  = lane & 15;

    int gw  = blockIdx.x * 4 + wv;
    int nqt = T >> 4;
    int qt  = gw % nqt;
    int bh  = gw / nqt;
    if (bh >= BH) return;
    int q0 = qt << 4;

    const bf16* Q  = Qg + (size_t)bh * T * 64;
    const bf16* Kp = Kg + (size_t)bh * T * 64;
    const bf16* Vp = Vt + (size_t)bh * 64 * T;

    v16bf aq[2];
    #pragma unroll
    for (int ks = 0; ks < 2; ++ks) {
        int kb = half * 16 + ks * 32;
        const bf16* p = Q + (size_t)(q0 + l16) * 64 + kb;
        union { v16bf v; v8bf h[2]; } u;
        u.h[0] = *(const v8bf*)p;
        u.h[1] = *(const v8bf*)(p + 8);
        aq[ks] = u.v;
    }

    v8f zero8 = {0.f, 0.f, 0.f, 0.f, 0.f, 0.f, 0.f, 0.f};
    float mrow[8], lrow[8], alpha[8];
    v8f oacc[4];
    #pragma unroll
    for (int r = 0; r < 8; ++r) { mrow[r] = -1e30f; lrow[r] = 0.f; }
    #pragma unroll
    for (int f = 0; f < 4; ++f) oacc[f] = zero8;

    const float scl = 0.125f;  // 1/sqrt(64)

    for (int kt0 = 0; kt0 <= q0 + 15; kt0 += 32) {
        v8f sfr[2];
        #pragma unroll
        for (int ni = 0; ni < 2; ++ni) {
            int key  = kt0 + ni * 16 + l16;
            int keyc = key < T ? key : T - 1;
            sfr[ni] = zero8;
            #pragma unroll
            for (int ks = 0; ks < 2; ++ks) {
                int kb = half * 16 + ks * 32;
                const bf16* p = Kp + (size_t)keyc * 64 + kb;
                union { v16bf v; v8bf h[2]; } u;
                u.h[0] = *(const v8bf*)p;
                u.h[1] = *(const v8bf*)(p + 8);
                sfr[ni] = __builtin_amdgcn_wmma_f32_16x16x32_bf16(
                    false, aq[ks], false, u.v, (short)0, sfr[ni], false, false);
            }
        }
        #pragma unroll
        for (int r = 0; r < 8; ++r) {
            int qrow = q0 + r + half * 8;
            float v0 = sfr[0][r] * scl;
            float v1 = sfr[1][r] * scl;
            if (kt0 + l16 > qrow)       v0 = -1e30f;
            if (kt0 + 16 + l16 > qrow)  v1 = -1e30f;
            float mx = fmaxf(v0, v1);
            #pragma unroll
            for (int off = 1; off < 16; off <<= 1) mx = fmaxf(mx, __shfl_xor(mx, off, 32));
            float mn = fmaxf(mrow[r], mx);
            float a  = __expf(mrow[r] - mn);
            float p0 = __expf(v0 - mn);
            float p1 = __expf(v1 - mn);
            float sm = p0 + p1;
            #pragma unroll
            for (int off = 1; off < 16; off <<= 1) sm += __shfl_xor(sm, off, 32);
            lrow[r] = lrow[r] * a + sm;
            mrow[r] = mn;
            alpha[r] = a;
            Pst[wv][r + half * 8][l16]      = f2bf(p0);
            Pst[wv][r + half * 8][16 + l16] = f2bf(p1);
        }
        #pragma unroll
        for (int f = 0; f < 4; ++f)
            #pragma unroll
            for (int r = 0; r < 8; ++r) oacc[f][r] *= alpha[r];

        // Same-wave LDS is in-order (ISA sect. 7.3); stop compiler reordering only.
        asm volatile("" ::: "memory");

        v16bf ap;
        {
            int kb = half * 16;
            union { v16bf v; v8bf h[2]; } u;
            u.h[0] = *(v8bf*)&Pst[wv][l16][kb];
            u.h[1] = *(v8bf*)&Pst[wv][l16][kb + 8];
            ap = u.v;
        }
        #pragma unroll
        for (int f = 0; f < 4; ++f) {
            int dk = f * 16 + l16;
            int tb = kt0 + half * 16;
            if (tb > T - 16) tb = T - 16;
            const bf16* p = Vp + (size_t)dk * T + tb;
            union { v16bf v; v8bf h[2]; } u;
            u.h[0] = *(const v8bf*)p;
            u.h[1] = *(const v8bf*)(p + 8);
            oacc[f] = __builtin_amdgcn_wmma_f32_16x16x32_bf16(
                false, ap, false, u.v, (short)0, oacc[f], false, false);
        }
    }

    int b  = bh / H;
    int hh = bh - b * H;
    int D  = H * 64;
    float inv[8];
    #pragma unroll
    for (int r = 0; r < 8; ++r) inv[r] = 1.f / lrow[r];
    #pragma unroll
    for (int f = 0; f < 4; ++f)
        #pragma unroll
        for (int r = 0; r < 8; ++r) {
            float v = oacc[f][r] * inv[r];
            size_t row = (size_t)(b * T + q0 + r + half * 8);
            out[row * D + hh * 64 + f * 16 + l16] = f2bf(v);
        }
}

// ---------------------------------------------------------------------------
// Host-side launcher
// ---------------------------------------------------------------------------
extern "C" void kernel_launch(void* const* d_in, const int* in_sizes, int n_in,
                              void* d_out, int out_size, void* d_ws, size_t ws_size,
                              hipStream_t stream) {
    (void)in_sizes; (void)n_in; (void)out_size; (void)ws_size;
    constexpr int Bb = 4, T = 2048, D = 1024, H = 16;
    constexpr int M = Bb * T;  // 8192
    constexpr size_t GEMM_LDS = 4 * 128 * 40 * sizeof(bf16);  // 40960 B

    const float* x      = (const float*)d_in[0];
    const float* ln1_g  = (const float*)d_in[1];
    const float* ln1_b  = (const float*)d_in[2];
    const float* qkv_w  = (const float*)d_in[3];
    const float* qkv_b  = (const float*)d_in[4];
    const float* proj_w = (const float*)d_in[5];
    const float* proj_b = (const float*)d_in[6];
    const float* ln2_g  = (const float*)d_in[7];
    const float* ln2_b  = (const float*)d_in[8];
    const float* fc1_w  = (const float*)d_in[9];
    const float* fc1_b  = (const float*)d_in[10];
    const float* fc2_w  = (const float*)d_in[11];
    const float* fc2_b  = (const float*)d_in[12];
    float* out = (float*)d_out;

    char* ws = (char*)d_ws;
    size_t off = 0;
    auto alloc = [&](size_t bytes) {
        void* p = ws + off;
        off = (off + bytes + 255) & ~(size_t)255;
        return p;
    };
    bf16* wt_qkv  = (bf16*)alloc((size_t)D * 3 * D * 2);   // [3D, D]   (N,K)
    bf16* wt_proj = (bf16*)alloc((size_t)D * D * 2);       // [D, D]
    bf16* wt_fc1  = (bf16*)alloc((size_t)D * 4 * D * 2);   // [4D, D]
    bf16* wt_fc2  = (bf16*)alloc((size_t)4 * D * D * 2);   // [D, 4D]
    bf16* ln1_out = (bf16*)alloc((size_t)M * D * 2);
    bf16* qbuf    = (bf16*)alloc((size_t)M * D * 2);
    bf16* kbuf    = (bf16*)alloc((size_t)M * D * 2);
    bf16* vtbuf   = (bf16*)alloc((size_t)M * D * 2);
    bf16* attn_o  = (bf16*)alloc((size_t)M * D * 2);
    float* x2     = (float*)alloc((size_t)M * D * 4);
    bf16* ln2_out = (bf16*)alloc((size_t)M * D * 2);
    bf16* hbuf    = (bf16*)alloc((size_t)M * 4 * D * 2);

    // 1) weights -> bf16, transposed to [N][K]
    transpose_bf16_kernel<<<dim3(3 * D / 32, D / 32), 256, 0, stream>>>(qkv_w,  wt_qkv,  D,     3 * D);
    transpose_bf16_kernel<<<dim3(D / 32,     D / 32), 256, 0, stream>>>(proj_w, wt_proj, D,     D);
    transpose_bf16_kernel<<<dim3(4 * D / 32, D / 32), 256, 0, stream>>>(fc1_w,  wt_fc1,  D,     4 * D);
    transpose_bf16_kernel<<<dim3(D / 32, 4 * D / 32), 256, 0, stream>>>(fc2_w,  wt_fc2,  4 * D, D);

    // 2) ln1
    ln_bf16_kernel<<<M, 256, 0, stream>>>(x, ln1_g, ln1_b, ln1_out, D);
    // 3) qkv GEMM (scatter epilogue -> Q, K, V^T)
    gemm_bf16_kernel<<<dim3(3 * D / 128, M / 128), 128, GEMM_LDS, stream>>>(
        ln1_out, wt_qkv, qkv_b, M, 3 * D, D, GMODE_QKV,
        nullptr, nullptr, nullptr, qbuf, kbuf, vtbuf, T, H);
    // 4) flash attention
    {
        int waves = Bb * H * (T / 16);
        attn_kernel<<<waves / 4, 128, 0, stream>>>(qbuf, kbuf, vtbuf, attn_o, H, T, Bb * H);
    }
    // 5) proj GEMM + residual -> x2 (fp32)
    gemm_bf16_kernel<<<dim3(D / 128, M / 128), 128, GEMM_LDS, stream>>>(
        attn_o, wt_proj, proj_b, M, D, D, GMODE_RES,
        x, x2, nullptr, nullptr, nullptr, nullptr, T, H);
    // 6) ln2
    ln_bf16_kernel<<<M, 256, 0, stream>>>(x2, ln2_g, ln2_b, ln2_out, D);
    // 7) fc1 GEMM + exact GELU -> h (bf16)
    gemm_bf16_kernel<<<dim3(4 * D / 128, M / 128), 128, GEMM_LDS, stream>>>(
        ln2_out, wt_fc1, fc1_b, M, 4 * D, D, GMODE_GELU,
        nullptr, nullptr, hbuf, nullptr, nullptr, nullptr, T, H);
    // 8) fc2 GEMM + residual -> out (fp32)
    gemm_bf16_kernel<<<dim3(D / 128, M / 128), 128, GEMM_LDS, stream>>>(
        hbuf, wt_fc2, fc2_b, M, D, 4 * D, GMODE_RES,
        x2, out, nullptr, nullptr, nullptr, nullptr, T, H);
}
